// ConnectionNetwork_81492709474821
// MI455X (gfx1250) — matrix operand order repaired and checked by reading, not compiled, not run on hardware
//
#include <hip/hip_runtime.h>
#include <hip/hip_bf16.h>

// ---------------------------------------------------------------------------
// ConnectionNetwork on MI455X (gfx1250, wave32)
//   1) f32 -> f16 convert of descriptors + W1 matrices
//   2) WMMA (v_wmma_f32_16x16x32_f16) GEMMs: A_cw, B_cw, A_ccw, B_ccw
//   3) Pair-score kernel (LDS b128 path): K = exp(S_cw + S_ccw^T), plus K^T
//   4) Sinkhorn via scaling vectors: u = 1/(K v), v = 1/(K^T u), 100 iters
//      (float4 / global_load_b128 matvecs, all resident in 192MB L2)
//   5) P = diag(u) K diag(v), in place on d_out (float4 RMW)
// ---------------------------------------------------------------------------

typedef __attribute__((ext_vector_type(16))) _Float16 v16h;
typedef __attribute__((ext_vector_type(8)))  _Float16 v8h;
typedef __attribute__((ext_vector_type(8)))  float    v8f;

#define NN 1024
#define DD 128

// ---------------- 1) f32 -> f16 convert ----------------
__global__ void cvt_f32_to_f16(const float* __restrict__ in,
                               _Float16* __restrict__ out, int n) {
  int idx = blockIdx.x * 256 + threadIdx.x;
  if (idx < n) out[idx] = (_Float16)in[idx];
}

// ---------------- 2) WMMA GEMM ----------------
// out[q][n][j] = sum_d desc[n][d] * W1[j][colOff+d]   (W1 is [128 x 256])
// q=0: W1a_cw  q=1: W1b_cw  q=2: W1a_ccw  q=3: W1b_ccw
// One wave computes one 16x16 tile; K=128 in 4 WMMA steps of 32.
__global__ void gemm_ab_wmma(const _Float16* __restrict__ descH,   // [1024,128]
                             const _Float16* __restrict__ WcwH,    // [128,256]
                             const _Float16* __restrict__ WccwH,   // [128,256]
                             float* __restrict__ outAB) {          // [4,1024,128]
  const int wave   = threadIdx.x >> 5;
  const int lane   = threadIdx.x & 31;
  const int tileId = blockIdx.x * 8 + wave;   // 2048 tiles total
  const int q  = tileId >> 9;                 // 0..3
  const int t  = tileId & 511;
  const int ti = t >> 3;                      // 64 row tiles
  const int tj = t & 7;                       // 8  col tiles
  const _Float16* __restrict__ W = (q < 2) ? WcwH : WccwH;
  const int colOff = (q & 1) ? DD : 0;
  const int m  = lane & 15;                   // row (A/D) or col (B)
  const int g  = lane >> 4;                   // lane half-group
  const int i0 = ti * 16, j0 = tj * 16;

  const v8h* __restrict__ pa = (const v8h*)(descH + (size_t)(i0 + m) * DD);
  const v8h* __restrict__ pb = (const v8h*)(W + (size_t)(j0 + m) * (2 * DD) + colOff);

  v8f c = {};
  #pragma unroll
  for (int kb = 0; kb < DD; kb += 32) {
    // A fragment (ISA 7.12.2, 16-bit A 16x32): lane group g holds
    // K = [kb+8g .. kb+8g+7] and [kb+16+8g .. kb+16+8g+7]  -> two 16B loads
    v8h alo = pa[(kb + 8 * g) >> 3];
    v8h ahi = pa[(kb + 16 + 8 * g) >> 3];
    // B fragment (32x16): lane group g holds K = [kb+8g .. kb+8g+7] and [kb+16+8g .. kb+16+8g+7]
    v8h blo = pb[(kb + 8 * g) >> 3];
    v8h bhi = pb[(kb + 16 + 8 * g) >> 3];
    v16h a, b;
    #pragma unroll
    for (int h = 0; h < 8; ++h) {
      a[h] = alo[h]; a[h + 8] = ahi[h];
      b[h] = blo[h]; b[h + 8] = bhi[h];
    }
    c = __builtin_amdgcn_wmma_f32_16x16x32_f16(
        /*neg_a=*/false, a, /*neg_b=*/false, b,
        /*c_mod=*/(short)0, c, /*reuse_a=*/false, /*reuse_b=*/false);
  }
  // D layout: VGPR r -> row r + 8g, col = lane&15
  float* __restrict__ out = outAB + (size_t)q * NN * DD;
  #pragma unroll
  for (int r = 0; r < 8; ++r)
    out[(size_t)(i0 + r + 8 * g) * DD + (j0 + m)] = c[r];
}

// ---------------- 3) Pair scores -> K = exp(S), plus K^T ----------------
// S[i,j] = b2cw + sum_d relu(Acw[i,d]+Bcw[j,d]+b1cw[d])*w2cw[d]
//        + b2ccw + sum_d relu(Accw[j,d]+Bccw[i,d]+b1ccw[d])*w2ccw[d]
// diag(S)=0 -> K[i,i]=1
// LDS pitch 132 floats: 132 % 64 = 4 -> the 16 lanes of a half-wave hit
// distinct banks, and rows stay 16B-aligned so the loop runs on ds_load_b128.
#define LP 132
__global__ void score_exp_kernel(const float* __restrict__ AB,
                                 const float* __restrict__ b1cw,
                                 const float* __restrict__ w2cw,
                                 const float* __restrict__ b2cw,
                                 const float* __restrict__ b1ccw,
                                 const float* __restrict__ w2ccw,
                                 const float* __restrict__ b2ccw,
                                 float* __restrict__ K,
                                 float* __restrict__ KT) {
  __shared__ __align__(16) float sAcw[16 * LP], sBcw[16 * LP];
  __shared__ __align__(16) float sAccw[16 * LP], sBccw[16 * LP];
  __shared__ __align__(16) float sb1cw[DD], sw2cw[DD], sb1ccw[DD], sw2ccw[DD];

  const int tx = threadIdx.x, ty = threadIdx.y;
  const int id = ty * 16 + tx;
  const int i0 = blockIdx.y * 16, j0 = blockIdx.x * 16;

  const float4* Acw4  = (const float4*)(AB + (size_t)0 * NN * DD + (size_t)i0 * DD);
  const float4* Bcw4  = (const float4*)(AB + (size_t)1 * NN * DD + (size_t)j0 * DD);
  const float4* Accw4 = (const float4*)(AB + (size_t)2 * NN * DD + (size_t)j0 * DD);
  const float4* Bccw4 = (const float4*)(AB + (size_t)3 * NN * DD + (size_t)i0 * DD);

  // 16 rows x 128 cols = 512 float4 per tile; 256 threads -> 2 each
  #pragma unroll
  for (int k = 0; k < 2; ++k) {
    int idx4 = id + k * 256;
    int r = idx4 >> 5, c4 = idx4 & 31;           // 32 float4 per row
    *(float4*)&sAcw [r * LP + c4 * 4] = Acw4 [idx4];
    *(float4*)&sBcw [r * LP + c4 * 4] = Bcw4 [idx4];
    *(float4*)&sAccw[r * LP + c4 * 4] = Accw4[idx4];
    *(float4*)&sBccw[r * LP + c4 * 4] = Bccw4[idx4];
  }
  if (id < DD) {
    sb1cw[id]  = b1cw[id];  sw2cw[id]  = w2cw[id];
    sb1ccw[id] = b1ccw[id]; sw2ccw[id] = w2ccw[id];
  }
  __syncthreads();

  const int i = i0 + ty, j = j0 + tx;
  float acw = 0.f, accw = 0.f;
  #pragma unroll 8
  for (int d4 = 0; d4 < DD / 4; ++d4) {
    float4 a1 = *(const float4*)&sAcw [ty * LP + d4 * 4];
    float4 b1 = *(const float4*)&sBcw [tx * LP + d4 * 4];
    float4 a2 = *(const float4*)&sAccw[tx * LP + d4 * 4];
    float4 b2 = *(const float4*)&sBccw[ty * LP + d4 * 4];
    float4 c1 = *(const float4*)&sb1cw [d4 * 4];
    float4 w1 = *(const float4*)&sw2cw [d4 * 4];
    float4 c2 = *(const float4*)&sb1ccw[d4 * 4];
    float4 w2 = *(const float4*)&sw2ccw[d4 * 4];
    acw  = fmaf(fmaxf(a1.x + b1.x + c1.x, 0.f), w1.x, acw);
    acw  = fmaf(fmaxf(a1.y + b1.y + c1.y, 0.f), w1.y, acw);
    acw  = fmaf(fmaxf(a1.z + b1.z + c1.z, 0.f), w1.z, acw);
    acw  = fmaf(fmaxf(a1.w + b1.w + c1.w, 0.f), w1.w, acw);
    accw = fmaf(fmaxf(a2.x + b2.x + c2.x, 0.f), w2.x, accw);
    accw = fmaf(fmaxf(a2.y + b2.y + c2.y, 0.f), w2.y, accw);
    accw = fmaf(fmaxf(a2.z + b2.z + c2.z, 0.f), w2.z, accw);
    accw = fmaf(fmaxf(a2.w + b2.w + c2.w, 0.f), w2.w, accw);
  }
  float s  = acw + b2cw[0] + accw + b2ccw[0];
  float kv = (i == j) ? 1.0f : __expf(s);   // diag: exp(0) = 1
  K [(size_t)i * NN + j] = kv;
  KT[(size_t)j * NN + i] = kv;
}

// ---------------- 4) Sinkhorn matvec:  y = 1 ./ (M x) ----------------
// 1024 blocks (one row each) x 128 threads; float4 / b128 loads, L2-resident.
__global__ void matvec_inv(const float* __restrict__ M,
                           const float* __restrict__ x,
                           float* __restrict__ y) {
  __shared__ float red[128];
  const int row = blockIdx.x;
  const float4* __restrict__ r4 = (const float4*)(M + (size_t)row * NN);
  const float4* __restrict__ x4 = (const float4*)x;
  float s = 0.f;
  #pragma unroll
  for (int k = 0; k < 2; ++k) {
    int j = threadIdx.x + k * 128;          // 256 float4 per row
    float4 a = r4[j], b = x4[j];
    s = fmaf(a.x, b.x, s);
    s = fmaf(a.y, b.y, s);
    s = fmaf(a.z, b.z, s);
    s = fmaf(a.w, b.w, s);
  }
  red[threadIdx.x] = s;
  __syncthreads();
  #pragma unroll
  for (int off = 64; off > 0; off >>= 1) {
    if (threadIdx.x < off) red[threadIdx.x] += red[threadIdx.x + off];
    __syncthreads();
  }
  if (threadIdx.x == 0) y[row] = 1.0f / red[0];
}

__global__ void set_ones(float* __restrict__ v, int n) {
  int idx = blockIdx.x * 256 + threadIdx.x;
  if (idx < n) v[idx] = 1.0f;
}

// ---------------- 5) P = diag(u) K diag(v), in place (float4) ----------------
__global__ void scale_uv(float* __restrict__ P,
                         const float* __restrict__ u,
                         const float* __restrict__ v) {
  int idx4 = blockIdx.x * 256 + threadIdx.x;  // 256K float4
  int i  = idx4 >> 8;                         // 256 float4 per row
  int j4 = idx4 & 255;
  float  ui = u[i];
  float4 vv = ((const float4*)v)[j4];
  float4 p  = ((float4*)P)[idx4];
  p.x *= ui * vv.x; p.y *= ui * vv.y; p.z *= ui * vv.z; p.w *= ui * vv.w;
  ((float4*)P)[idx4] = p;
}

// ---------------------------------------------------------------------------
extern "C" void kernel_launch(void* const* d_in, const int* in_sizes, int n_in,
                              void* d_out, int out_size, void* d_ws, size_t ws_size,
                              hipStream_t stream) {
  const float* desc   = (const float*)d_in[0];  // [1024,128]
  const float* W1_cw  = (const float*)d_in[1];  // [128,256]
  const float* b1_cw  = (const float*)d_in[2];  // [128]
  const float* w2_cw  = (const float*)d_in[3];  // [128]
  const float* b2_cw  = (const float*)d_in[4];  // [1]
  const float* W1_ccw = (const float*)d_in[5];
  const float* b1_ccw = (const float*)d_in[6];
  const float* w2_ccw = (const float*)d_in[7];
  const float* b2_ccw = (const float*)d_in[8];

  char* ws = (char*)d_ws;
  _Float16* descH = (_Float16*)(ws);                           // 256 KB
  _Float16* WcwH  = (_Float16*)(ws + (256 << 10));             //  64 KB
  _Float16* WccwH = (_Float16*)(ws + (320 << 10));             //  64 KB
  float*    AB    = (float*)   (ws + (384 << 10));             //   2 MB
  float*    KT    = (float*)   (ws + (384 << 10) + (2 << 20)); //   4 MB
  float*    u     = (float*)   (ws + (384 << 10) + (6 << 20));
  float*    v     = (float*)   (ws + (384 << 10) + (6 << 20) + 4096);
  float*    K     = (float*)d_out;                             // K lives in d_out

  // 1) convert to f16
  cvt_f32_to_f16<<<(NN * DD + 255) / 256, 256, 0, stream>>>(desc, descH, NN * DD);
  cvt_f32_to_f16<<<(DD * 2 * DD + 255) / 256, 256, 0, stream>>>(W1_cw, WcwH, DD * 2 * DD);
  cvt_f32_to_f16<<<(DD * 2 * DD + 255) / 256, 256, 0, stream>>>(W1_ccw, WccwH, DD * 2 * DD);

  // 2) four GEMMs via WMMA: 2048 tiles, 8 waves/block
  gemm_ab_wmma<<<256, 256, 0, stream>>>(descH, WcwH, WccwH, AB);

  // 3) K = exp(S), K^T
  score_exp_kernel<<<dim3(NN / 16, NN / 16), dim3(16, 16), 0, stream>>>(
      AB, b1_cw, w2_cw, b2_cw, b1_ccw, w2_ccw, b2_ccw, K, KT);

  // 4) Sinkhorn on scaling vectors
  set_ones<<<(NN + 255) / 256, 256, 0, stream>>>(v, NN);
  for (int it = 0; it < 100; ++it) {
    matvec_inv<<<NN, 128, 0, stream>>>(K, v, u);   // u = 1/(K v)
    matvec_inv<<<NN, 128, 0, stream>>>(KT, u, v);  // v = 1/(K^T u)
  }

  // 5) P = diag(u) K diag(v)
  scale_uv<<<(NN * NN / 4 + 255) / 256, 256, 0, stream>>>(K, u, v);
}
